// GraphAttentionEmbedding_20684562498291
// MI455X (gfx1250) — matrix operand order repaired and checked by reading, not compiled, not run on hardware
//
#include <hip/hip_runtime.h>
#include <hip/hip_bf16.h>

#define N_NODES 50000
#define N_EDGES 500000

typedef __bf16 bf16_t;
typedef __attribute__((ext_vector_type(16))) __bf16 v16bf;
typedef __attribute__((ext_vector_type(8)))  __bf16 v8bf;
typedef __attribute__((ext_vector_type(8)))  float  v8f;

// ---------------- WMMA helpers (per CDNA5 ISA layouts) ----------------

static __device__ inline v8f wmma_bf16f32(v16bf a, v16bf b, v8f c) {
  // v_wmma_f32_16x16x32_bf16: (neg_a, A, neg_b, B, c_mod, C, reuse_a, reuse_b)
  return __builtin_amdgcn_wmma_f32_16x16x32_bf16(false, a, false, b, (short)0, c, false, false);
}

// A fragment (16x32 bf16): lane half=lane>>4; K chunks [kb+8*half, +8) and [kb+16+8*half, +8)
static __device__ inline v16bf load_frag_A(const bf16_t* __restrict__ row, int kb, int half) {
  v8bf lo = *(const v8bf*)(row + kb + 8 * half);
  v8bf hi = *(const v8bf*)(row + kb + 16 + 8 * half);
  v16bf r;
#pragma unroll
  for (int i = 0; i < 8; ++i) { r[i] = lo[i]; r[8 + i] = hi[i]; }
  return r;
}

// B fragment (32x16 bf16): lane holds column; K chunks [kb+8*half, +8) and [kb+16+8*half, +8).
// "row" points at row n of pre-transposed Wt[NC][K] (== column n of W[K][NC]).
static __device__ inline v16bf load_frag_B(const bf16_t* __restrict__ row, int kb, int half) {
  v8bf lo = *(const v8bf*)(row + kb + 8 * half);
  v8bf hi = *(const v8bf*)(row + kb + 16 + 8 * half);
  v16bf r;
#pragma unroll
  for (int i = 0; i < 8; ++i) { r[i] = lo[i]; r[8 + i] = hi[i]; }
  return r;
}

// Build an edge_attr A-fragment in registers: K<64 -> cos(rel*tw+tb), K>=64 -> msg[K-64].
// kb in {0,32,64,96}, so the branch is uniform per k-step.
static __device__ inline v16bf edge_frag(int kb, int half, float rel,
                                         const float* __restrict__ msgRow,
                                         const float* __restrict__ tw,
                                         const float* __restrict__ tb) {
  v16bf r;
  int k0 = kb + 8 * half;
  int k1 = kb + 16 + 8 * half;
  if (kb < 64) {
#pragma unroll
    for (int j = 0; j < 8; ++j) r[j]     = (bf16_t)__cosf(rel * tw[k0 + j] + tb[k0 + j]);
#pragma unroll
    for (int j = 0; j < 8; ++j) r[8 + j] = (bf16_t)__cosf(rel * tw[k1 + j] + tb[k1 + j]);
  } else {
#pragma unroll
    for (int j = 0; j < 8; ++j) r[j]     = (bf16_t)msgRow[k0 - 64 + j];
#pragma unroll
    for (int j = 0; j < 8; ++j) r[8 + j] = (bf16_t)msgRow[k1 - 64 + j];
  }
  return r;
}

// Order-preserving float<->uint map for atomic segment-max.
static __device__ inline unsigned f2ord(float f) {
  unsigned u = __float_as_uint(f);
  return (u & 0x80000000u) ? ~u : (u | 0x80000000u);
}
static __device__ inline float ord2f(unsigned u) {
  unsigned v = (u & 0x80000000u) ? (u & 0x7fffffffu) : ~u;
  return __uint_as_float(v);
}

// ---------------- prep / elementwise kernels ----------------

__global__ void fill_f32(float* p, float v, int n) {
  int i = blockIdx.x * blockDim.x + threadIdx.x;
  if (i < n) p[i] = v;
}
__global__ void fill_u32(unsigned* p, unsigned v, int n) {
  int i = blockIdx.x * blockDim.x + threadIdx.x;
  if (i < n) p[i] = v;
}
__global__ void to_bf16_kernel(const float* __restrict__ x, bf16_t* __restrict__ y, int n) {
  int i = blockIdx.x * blockDim.x + threadIdx.x;
  if (i < n) y[i] = (bf16_t)x[i];
}
// W[K][NC] row-major -> Wt[rowOff+n][K] bf16 (transposed)
__global__ void transpose_w_bf16(const float* __restrict__ W, bf16_t* __restrict__ Wt,
                                 int K, int NC, int rowOff) {
  int idx = blockIdx.x * blockDim.x + threadIdx.x;
  if (idx >= K * NC) return;
  int n = idx / K, k = idx - n * K;
  Wt[(size_t)(rowOff + n) * K + k] = (bf16_t)W[(size_t)k * NC + n];
}
__global__ void copy_f32(const float* __restrict__ a, float* __restrict__ b, int n, int off) {
  int i = blockIdx.x * blockDim.x + threadIdx.x;
  if (i < n) b[off + i] = a[i];
}

// ---------------- GEMM: C[M,NC] = A_bf16[M,K] @ Wt_bf16[NC,K] + bias ----------------
// grid.x = M/16, grid.y * 8 waves cover NC/16 tiles; one 16x16 C tile per wave.
__global__ void gemm_bf16_kernel(const bf16_t* __restrict__ A, const bf16_t* __restrict__ Wt,
                                 const float* __restrict__ bias, float* __restrict__ C,
                                 int K, int NC) {
  int lane = threadIdx.x & 31;
  int wave = threadIdx.x >> 5;
  int half = lane >> 4, r = lane & 15;
  int tileN = blockIdx.y * (blockDim.x >> 5) + wave;
  if (tileN * 16 >= NC) return;
  int rowBase = blockIdx.x * 16;
  const bf16_t* aRow = A + (size_t)(rowBase + r) * K;
  const bf16_t* bRow = Wt + (size_t)(tileN * 16 + r) * K;
  v8f acc = {};
  for (int kb = 0; kb < K; kb += 32)
    acc = wmma_bf16f32(load_frag_A(aRow, kb, half), load_frag_B(bRow, kb, half), acc);
  float bv = bias[tileN * 16 + r];
#pragma unroll
  for (int i = 0; i < 8; ++i) {
    int row = rowBase + i + 8 * half;           // C layout: VGPR i -> row i+8*half, lane -> col r
    C[(size_t)row * NC + tileN * 16 + r] = acc[i] + bv;
  }
}

// ---------------- edge pass 1: scores + segment-max ----------------
// Each wave handles 16 edges; per head, e-tile = one WMMA C tile; score = q[dst] . (k[src]+e) / 4.
__global__ void edge_scores_kernel(const int* __restrict__ src, const int* __restrict__ dst,
    const float* __restrict__ lu, const float* __restrict__ tvec,
    const float* __restrict__ msg, const float* __restrict__ tw, const float* __restrict__ tb,
    const bf16_t* __restrict__ ewt, const float* __restrict__ proj, int PS, int qOff, int kOff,
    int H, float* __restrict__ scores, unsigned* __restrict__ mord, int E) {
  int lane = threadIdx.x & 31;
  int wave = threadIdx.x >> 5;
  int half = lane >> 4, r = lane & 15;
  int eBase = (blockIdx.x * (blockDim.x >> 5) + wave) * 16;
  if (eBase >= E) return;

  int eRow = eBase + r;                          // A-fragment row for this lane
  float rel = lu[src[eRow]] - tvec[eRow];
  const float* msgRow = msg + (size_t)eRow * 64;
  v16bf af[4];
#pragma unroll
  for (int kt = 0; kt < 4; ++kt) af[kt] = edge_frag(kt * 32, half, rel, msgRow, tw, tb);

  int srcs[8], dsts[8];                          // C-tile rows: edges i + 8*half
#pragma unroll
  for (int i = 0; i < 8; ++i) {
    int e2 = eBase + i + 8 * half;
    srcs[i] = src[e2]; dsts[i] = dst[e2];
  }

  for (int h = 0; h < H; ++h) {
    const bf16_t* bRow = ewt + (size_t)(h * 16 + r) * 128;
    v8f acc = {};
#pragma unroll
    for (int kt = 0; kt < 4; ++kt)
      acc = wmma_bf16f32(af[kt], load_frag_B(bRow, kt * 32, half), acc);

    float sums[8];
#pragma unroll
    for (int i = 0; i < 8; ++i) {
      float kv = proj[(size_t)srcs[i] * PS + kOff + h * 16 + r] + acc[i];
      float qv = proj[(size_t)dsts[i] * PS + qOff + h * 16 + r];
      float s = kv * qv;
      s += __shfl_xor(s, 1, 32);                 // masks <16 keep reduction inside each half
      s += __shfl_xor(s, 2, 32);
      s += __shfl_xor(s, 4, 32);
      s += __shfl_xor(s, 8, 32);
      sums[i] = s;
    }
    if (r < 8) {
      float sc = sums[0];
#pragma unroll
      for (int i = 1; i < 8; ++i) if (r == i) sc = sums[i];
      sc *= 0.25f;                               // 1/sqrt(16)
      int e2 = eBase + 8 * half + r;
      scores[(size_t)e2 * H + h] = sc;
      atomicMax(&mord[(size_t)dst[e2] * H + h], f2ord(sc));
    }
  }
}

// ---------------- edge pass 2: exp + segment-sum ----------------
__global__ void edge_softmax_kernel(const int* __restrict__ dst, float* __restrict__ scores,
                                    const unsigned* __restrict__ mord, float* __restrict__ denom,
                                    int H, int total) {
  int i = blockIdx.x * blockDim.x + threadIdx.x;
  if (i >= total) return;
  int e = i / H, h = i - e * H;
  int d = dst[e];
  float a = __expf(scores[i] - ord2f(mord[(size_t)d * H + h]));
  atomicAdd(&denom[(size_t)d * H + h], a);
  scores[i] = a;
}

// ---------------- edge pass 3: recompute e via WMMA, scatter-aggregate ----------------
__global__ void edge_aggregate_kernel(const int* __restrict__ src, const int* __restrict__ dst,
    const float* __restrict__ lu, const float* __restrict__ tvec,
    const float* __restrict__ msg, const float* __restrict__ tw, const float* __restrict__ tb,
    const bf16_t* __restrict__ ewt, const float* __restrict__ proj, int PS, int vOff, int H,
    const float* __restrict__ scores, const float* __restrict__ denom,
    float* __restrict__ agg, int E) {
  int lane = threadIdx.x & 31;
  int wave = threadIdx.x >> 5;
  int half = lane >> 4, r = lane & 15;
  int eBase = (blockIdx.x * (blockDim.x >> 5) + wave) * 16;
  if (eBase >= E) return;

  int eRow = eBase + r;
  float rel = lu[src[eRow]] - tvec[eRow];
  const float* msgRow = msg + (size_t)eRow * 64;
  v16bf af[4];
#pragma unroll
  for (int kt = 0; kt < 4; ++kt) af[kt] = edge_frag(kt * 32, half, rel, msgRow, tw, tb);

  int srcs[8], dsts[8];
#pragma unroll
  for (int i = 0; i < 8; ++i) {
    int e2 = eBase + i + 8 * half;
    srcs[i] = src[e2]; dsts[i] = dst[e2];
  }
  int AC = H * 16;                               // aggregated feature width

  for (int h = 0; h < H; ++h) {
    const bf16_t* bRow = ewt + (size_t)(h * 16 + r) * 128;
    v8f acc = {};
#pragma unroll
    for (int kt = 0; kt < 4; ++kt)
      acc = wmma_bf16f32(af[kt], load_frag_B(bRow, kt * 32, half), acc);
#pragma unroll
    for (int i = 0; i < 8; ++i) {
      int e2 = eBase + i + 8 * half;
      int d = dsts[i];
      float a = scores[(size_t)e2 * H + h];
      float coef = a / (denom[(size_t)d * H + h] + 1e-16f);
      float vv = proj[(size_t)srcs[i] * PS + vOff + h * 16 + r] + acc[i];
      atomicAdd(&agg[(size_t)d * AC + h * 16 + r], coef * vv);
    }
  }
}

// ---------------- node epilogues ----------------
__global__ void node_h_kernel(const float* __restrict__ agg1, const float* __restrict__ proj1,
                              bf16_t* __restrict__ hbf, int total) {
  int i = blockIdx.x * blockDim.x + threadIdx.x;
  if (i >= total) return;
  int n = i >> 7, c = i & 127;
  float v = agg1[i] + proj1[(size_t)n * 512 + 384 + c];   // + skip(s1)
  hbf[i] = (bf16_t)fmaxf(v, 0.f);                          // relu
}
__global__ void node_out_kernel(const float* __restrict__ agg2, const float* __restrict__ proj2,
                                float* __restrict__ out, int total) {
  int i = blockIdx.x * blockDim.x + threadIdx.x;
  if (i >= total) return;
  int n = i >> 4, c = i & 15;
  out[i] = fmaxf(agg2[i] + proj2[(size_t)n * 64 + 48 + c], 0.f);
}

// ---------------- host orchestration ----------------

extern "C" void kernel_launch(void* const* d_in, const int* in_sizes, int n_in,
                              void* d_out, int out_size, void* d_ws, size_t ws_size,
                              hipStream_t stream) {
  (void)in_sizes; (void)n_in; (void)out_size; (void)ws_size;
  const int N = N_NODES, E = N_EDGES;

  const float* x   = (const float*)d_in[0];
  const float* lu  = (const float*)d_in[1];
  const float* tv  = (const float*)d_in[2];
  const float* msg = (const float*)d_in[3];
  const int*   ei  = (const int*)d_in[4];
  const int* src = ei, * dst = ei + E;
  const float* tw  = (const float*)d_in[5];
  const float* tb  = (const float*)d_in[6];
  const float* q1w = (const float*)d_in[7],  *q1b = (const float*)d_in[8];
  const float* k1w = (const float*)d_in[9],  *k1b = (const float*)d_in[10];
  const float* v1w = (const float*)d_in[11], *v1b = (const float*)d_in[12];
  const float* e1w = (const float*)d_in[13];
  const float* s1w = (const float*)d_in[14], *s1b = (const float*)d_in[15];
  const float* q2w = (const float*)d_in[16], *q2b = (const float*)d_in[17];
  const float* k2w = (const float*)d_in[18], *k2b = (const float*)d_in[19];
  const float* v2w = (const float*)d_in[20], *v2b = (const float*)d_in[21];
  const float* e2w = (const float*)d_in[22];
  const float* s2w = (const float*)d_in[23], *s2b = (const float*)d_in[24];
  float* out = (float*)d_out;

  char* wp = (char*)d_ws;
  auto walloc = [&](size_t bytes) { void* p = (void*)wp; wp += (bytes + 255) & ~(size_t)255; return p; };
  bf16_t*   xbf   = (bf16_t*)  walloc((size_t)N * 128 * 2);
  float*    proj1 = (float*)   walloc((size_t)N * 512 * 4);   // [q1|k1|v1|s1]
  float*    sc1   = (float*)   walloc((size_t)E * 8 * 4);
  unsigned* m1    = (unsigned*)walloc((size_t)N * 8 * 4);
  float*    dn1   = (float*)   walloc((size_t)N * 8 * 4);
  float*    agg1  = (float*)   walloc((size_t)N * 128 * 4);
  bf16_t*   hbf   = (bf16_t*)  walloc((size_t)N * 128 * 2);
  float*    proj2 = (float*)   walloc((size_t)N * 64 * 4);    // [q2|k2|v2|s2]
  float*    sc2   = (float*)   walloc((size_t)E * 4);
  unsigned* m2    = (unsigned*)walloc((size_t)N * 4);
  float*    dn2   = (float*)   walloc((size_t)N * 4);
  float*    agg2  = (float*)   walloc((size_t)N * 16 * 4);
  bf16_t*   wt1   = (bf16_t*)  walloc(512 * 128 * 2);
  bf16_t*   ewt1  = (bf16_t*)  walloc(128 * 128 * 2);
  bf16_t*   wt2   = (bf16_t*)  walloc(64 * 128 * 2);
  bf16_t*   ewt2  = (bf16_t*)  walloc(16 * 128 * 2);
  float*    b1    = (float*)   walloc(512 * 4);
  float*    b2    = (float*)   walloc(64 * 4);

  // --- weight prep: transpose to [NC,K] bf16 ---
  transpose_w_bf16<<<64, 256, 0, stream>>>(q1w, wt1, 128, 128, 0);
  transpose_w_bf16<<<64, 256, 0, stream>>>(k1w, wt1, 128, 128, 128);
  transpose_w_bf16<<<64, 256, 0, stream>>>(v1w, wt1, 128, 128, 256);
  transpose_w_bf16<<<64, 256, 0, stream>>>(s1w, wt1, 128, 128, 384);
  transpose_w_bf16<<<64, 256, 0, stream>>>(e1w, ewt1, 128, 128, 0);
  transpose_w_bf16<<<8, 256, 0, stream>>>(q2w, wt2, 128, 16, 0);
  transpose_w_bf16<<<8, 256, 0, stream>>>(k2w, wt2, 128, 16, 16);
  transpose_w_bf16<<<8, 256, 0, stream>>>(v2w, wt2, 128, 16, 32);
  transpose_w_bf16<<<8, 256, 0, stream>>>(s2w, wt2, 128, 16, 48);
  transpose_w_bf16<<<8, 256, 0, stream>>>(e2w, ewt2, 128, 16, 0);
  copy_f32<<<1, 128, 0, stream>>>(q1b, b1, 128, 0);
  copy_f32<<<1, 128, 0, stream>>>(k1b, b1, 128, 128);
  copy_f32<<<1, 128, 0, stream>>>(v1b, b1, 128, 256);
  copy_f32<<<1, 128, 0, stream>>>(s1b, b1, 128, 384);
  copy_f32<<<1, 64, 0, stream>>>(q2b, b2, 16, 0);
  copy_f32<<<1, 64, 0, stream>>>(k2b, b2, 16, 16);
  copy_f32<<<1, 64, 0, stream>>>(v2b, b2, 16, 32);
  copy_f32<<<1, 64, 0, stream>>>(s2b, b2, 16, 48);

  // --- activations / init ---
  to_bf16_kernel<<<(N * 128 + 255) / 256, 256, 0, stream>>>(x, xbf, N * 128);
  fill_u32<<<(N * 8 + 255) / 256, 256, 0, stream>>>(m1, 0u, N * 8);
  fill_f32<<<(N * 8 + 255) / 256, 256, 0, stream>>>(dn1, 0.f, N * 8);
  fill_f32<<<(N * 128 + 255) / 256, 256, 0, stream>>>(agg1, 0.f, N * 128);
  fill_u32<<<(N + 255) / 256, 256, 0, stream>>>(m2, 0u, N);
  fill_f32<<<(N + 255) / 256, 256, 0, stream>>>(dn2, 0.f, N);
  fill_f32<<<(N * 16 + 255) / 256, 256, 0, stream>>>(agg2, 0.f, N * 16);

  // --- conv1 ---
  gemm_bf16_kernel<<<dim3(N / 16, 4), 256, 0, stream>>>(xbf, wt1, b1, proj1, 128, 512);

  int egrid = (E / 16 + 7) / 8;  // 16 edges per wave, 8 waves per block
  edge_scores_kernel<<<egrid, 256, 0, stream>>>(src, dst, lu, tv, msg, tw, tb,
      ewt1, proj1, 512, 0, 128, 8, sc1, m1, E);
  edge_softmax_kernel<<<(E * 8 + 255) / 256, 256, 0, stream>>>(dst, sc1, m1, dn1, 8, E * 8);
  edge_aggregate_kernel<<<egrid, 256, 0, stream>>>(src, dst, lu, tv, msg, tw, tb,
      ewt1, proj1, 512, 256, 8, sc1, dn1, agg1, E);
  node_h_kernel<<<(N * 128 + 255) / 256, 256, 0, stream>>>(agg1, proj1, hbf, N * 128);

  // --- conv2 ---
  gemm_bf16_kernel<<<dim3(N / 16, 1), 256, 0, stream>>>(hbf, wt2, b2, proj2, 128, 64);
  edge_scores_kernel<<<egrid, 256, 0, stream>>>(src, dst, lu, tv, msg, tw, tb,
      ewt2, proj2, 64, 0, 16, 1, sc2, m2, E);
  edge_softmax_kernel<<<(E + 255) / 256, 256, 0, stream>>>(dst, sc2, m2, dn2, 1, E);
  edge_aggregate_kernel<<<egrid, 256, 0, stream>>>(src, dst, lu, tv, msg, tw, tb,
      ewt2, proj2, 64, 32, 1, sc2, dn2, agg2, E);
  node_out_kernel<<<(N * 16 + 255) / 256, 256, 0, stream>>>(agg2, proj2, out, N * 16);
}
